// OTMatchingModule_3521873182812
// MI455X (gfx1250) — matrix operand order repaired and checked by reading, not compiled, not run on hardware
//
#include <hip/hip_runtime.h>
#include <math.h>

// Problem constants (from reference)
#define NF 4096
#define DF 512
#define EPSILON 0.05f
#define NUM_SINKHORN 50
#define NUM_FGW 10
#define EPS_LOG 1e-10f

static constexpr size_t N2 = (size_t)NF * (size_t)NF;

typedef __attribute__((ext_vector_type(2))) float v2f;
typedef __attribute__((ext_vector_type(8))) float v8f;
typedef __attribute__((ext_vector_type(8))) __bf16 bf8v;
typedef __attribute__((ext_vector_type(16))) __bf16 bf16v;

// FP32 WMMA: D = A(16x4)*B(4x16)+C  (exact path, K=512 feature GEMMs)
static __device__ __forceinline__ v8f wmma_f32(v2f a, v2f b, v8f c) {
  return __builtin_amdgcn_wmma_f32_16x16x4_f32(false, a, false, b, (short)0, c,
                                               false, false);
}
// BF16 WMMA: D = A(16x32)*B(32x16)+C, f32 accumulate (8x K per instruction)
static __device__ __forceinline__ v8f wmma_bf16(bf16v a, bf16v b, v8f c) {
  return __builtin_amdgcn_wmma_f32_16x16x32_bf16(false, a, false, b, (short)0,
                                                 c, false, false);
}

// A-fragment (16x32 bf16): lane(r,h) holds K = 8h+{0..7} and 16+8h+{0..7}
static __device__ __forceinline__ bf16v ldfragA(const __bf16* p /* row + kk + 8h */) {
  bf8v c0 = *(const bf8v*)(p);
  bf8v c1 = *(const bf8v*)(p + 16);
  return __builtin_shufflevector(c0, c1, 0, 1, 2, 3, 4, 5, 6, 7, 8, 9, 10, 11,
                                 12, 13, 14, 15);
}
// B-fragment (NT, row j of B): lane(r,h) holds K = 8h+{0..7} and 16+8h+{0..7}
static __device__ __forceinline__ bf16v ldfragB(const __bf16* p /* row + kk + 8h */) {
  bf8v c0 = *(const bf8v*)(p);
  bf8v c1 = *(const bf8v*)(p + 16);
  return __builtin_shufflevector(c0, c1, 0, 1, 2, 3, 4, 5, 6, 7, 8, 9, 10, 11,
                                 12, 13, 14, 15);
}

// ---------------------------------------------------------------------------
// Row squared-norms of (NF x DF)
// ---------------------------------------------------------------------------
__global__ __launch_bounds__(256) void k_rownorm(const float* __restrict__ X,
                                                 float* __restrict__ sq) {
  int row = blockIdx.x, t = threadIdx.x;
  const float* p = X + (size_t)row * DF;
  float s = 0.f;
  for (int j = t; j < DF; j += 256) { float x = p[j]; s += x * x; }
  __shared__ float sh[256];
  sh[t] = s; __syncthreads();
  for (int o = 128; o > 0; o >>= 1) { if (t < o) sh[t] += sh[t + o]; __syncthreads(); }
  if (t == 0) sq[row] = sh[0];
}

// ---------------------------------------------------------------------------
// Exact f32-WMMA NT GEMM + distance epilogue (K = DF = 512)
//   mode 0 : outF = max(0, |x|^2+|y|^2-2S)                      (C_W, f32)
//   mode 1 : d = sqrt(...); oH = bf16(d), oL = bf16(d - hi)     (D_gen/D_real)
// Wave tile 64x64 (4x4 frags); 8 waves (2x4) => 128x256 block tile.
// ---------------------------------------------------------------------------
__global__ __launch_bounds__(256) void k_gemm_nt_dist(
    const float* __restrict__ X, const float* __restrict__ Y,
    const float* __restrict__ sqx, const float* __restrict__ sqy,
    float* __restrict__ outF, __bf16* __restrict__ oH, __bf16* __restrict__ oL,
    int mode) {
  const int lane = threadIdx.x & 31, wave = threadIdx.x >> 5;
  const int wm = wave & 1, wn = wave >> 1;
  const int r = lane & 15, h = lane >> 4;
  const int ibase = blockIdx.x * 128 + wm * 64;
  const int jbase = blockIdx.y * 256 + wn * 64;

  v8f acc[4][4];
#pragma unroll
  for (int m = 0; m < 4; m++)
#pragma unroll
    for (int n = 0; n < 4; n++)
#pragma unroll
      for (int e = 0; e < 8; e++) acc[m][n][e] = 0.f;

  for (int kk = 0; kk < DF; kk += 4) {
    v2f af[4], bf[4];
#pragma unroll
    for (int m = 0; m < 4; m++)
      af[m] = *(const v2f*)(X + (size_t)(ibase + m * 16 + r) * DF + kk + 2 * h);
#pragma unroll
    for (int n = 0; n < 4; n++)
      bf[n] = *(const v2f*)(Y + (size_t)(jbase + n * 16 + r) * DF + kk + 2 * h);
#pragma unroll
    for (int m = 0; m < 4; m++)
#pragma unroll
      for (int n = 0; n < 4; n++)
        acc[m][n] = wmma_f32(af[m], bf[n], acc[m][n]);
  }

#pragma unroll
  for (int m = 0; m < 4; m++)
#pragma unroll
    for (int n = 0; n < 4; n++)
#pragma unroll
      for (int e = 0; e < 8; e++) {
        int i = ibase + m * 16 + e + 8 * h;
        int j = jbase + n * 16 + r;
        float d2 = fmaxf(sqx[i] + sqy[j] - 2.f * acc[m][n][e], 0.f);
        size_t idx = (size_t)i * NF + j;
        if (mode == 0) {
          outF[idx] = d2;
        } else {
          float d = sqrtf(d2);
          __bf16 hi = (__bf16)d;
          oH[idx] = hi;
          oL[idx] = (__bf16)(d - (float)hi);
        }
      }
}

// ---------------------------------------------------------------------------
// Split-bf16 NT GEMM (K = NF): C = (Ah+Al) * (Bh+Bl)^T, f32 accumulate,
// 3 WMMA passes (hh + hl + lh) ~ fp32 accuracy at 3/8 the f32-WMMA cost.
//   epi=0 : out = S                                    (T = D_gen @ P)
//   epi=1 : out = t1r[i]*t1c[j] + t2r[i]*t2c[j] - 2*S  (C_GW)
// Wave tile 64x32 (4x2 frags); 8 waves (2x4) => 128x128 block tile.
// ---------------------------------------------------------------------------
__global__ __launch_bounds__(256) void k_gemm_bf16split(
    const __bf16* __restrict__ Ah, const __bf16* __restrict__ Al,
    const __bf16* __restrict__ Bh, const __bf16* __restrict__ Bl,
    float* __restrict__ out,
    const float* __restrict__ t1r, const float* __restrict__ t1c,
    const float* __restrict__ t2r, const float* __restrict__ t2c, int epi) {
  const int lane = threadIdx.x & 31, wave = threadIdx.x >> 5;
  const int wm = wave & 1, wn = wave >> 1;       // 2x4 waves
  const int r = lane & 15, h = lane >> 4;
  const int ibase = blockIdx.x * 128 + wm * 64;  // 64 rows / wave
  const int jbase = blockIdx.y * 128 + wn * 32;  // 32 cols / wave

  v8f acc[4][2];
#pragma unroll
  for (int m = 0; m < 4; m++)
#pragma unroll
    for (int n = 0; n < 2; n++)
#pragma unroll
      for (int e = 0; e < 8; e++) acc[m][n][e] = 0.f;

#pragma unroll 2
  for (int kk = 0; kk < NF; kk += 32) {
    bf16v ah[4], al[4], bh[2], bl[2];
#pragma unroll
    for (int m = 0; m < 4; m++) {
      size_t off = (size_t)(ibase + m * 16 + r) * NF + kk + 8 * h;
      ah[m] = ldfragA(Ah + off);
      al[m] = ldfragA(Al + off);
    }
#pragma unroll
    for (int n = 0; n < 2; n++) {
      size_t off = (size_t)(jbase + n * 16 + r) * NF + kk + 8 * h;
      bh[n] = ldfragB(Bh + off);
      bl[n] = ldfragB(Bl + off);
    }
    if (kk + 256 < NF) {  // L2 prefetch a few K-steps ahead
      __builtin_prefetch((const void*)(Ah + (size_t)(ibase + r) * NF + kk + 256), 0, 1);
      __builtin_prefetch((const void*)(Bh + (size_t)(jbase + r) * NF + kk + 256), 0, 1);
    }
#pragma unroll
    for (int m = 0; m < 4; m++)
#pragma unroll
      for (int n = 0; n < 2; n++) {
        acc[m][n] = wmma_bf16(ah[m], bh[n], acc[m][n]);
        acc[m][n] = wmma_bf16(ah[m], bl[n], acc[m][n]);
        acc[m][n] = wmma_bf16(al[m], bh[n], acc[m][n]);
      }
  }

#pragma unroll
  for (int m = 0; m < 4; m++)
#pragma unroll
    for (int n = 0; n < 2; n++)
#pragma unroll
      for (int e = 0; e < 8; e++) {
        int i = ibase + m * 16 + e + 8 * h;
        int j = jbase + n * 16 + r;
        float s = acc[m][n][e];
        out[(size_t)i * NF + j] =
            epi ? (t1r[i] * t1c[j] + t2r[i] * t2c[j] - 2.f * s) : s;
      }
}

// ---------------------------------------------------------------------------
// f32 -> split bf16 (same layout)
// ---------------------------------------------------------------------------
__global__ __launch_bounds__(256) void k_cvt_split(const float* __restrict__ src,
                                                   __bf16* __restrict__ dh,
                                                   __bf16* __restrict__ dl) {
  for (size_t idx = (size_t)blockIdx.x * 256 + threadIdx.x; idx < N2;
       idx += (size_t)gridDim.x * 256) {
    float x = src[idx];
    __bf16 hi = (__bf16)x;
    dh[idx] = hi;
    dl[idx] = (__bf16)(x - (float)hi);
  }
}

// f32 -> split bf16, transposed (LDS 32x32 tiles, fully coalesced both sides)
__global__ __launch_bounds__(256) void k_cvt_split_T(const float* __restrict__ src,
                                                     __bf16* __restrict__ dh,
                                                     __bf16* __restrict__ dl) {
  __shared__ float tile[32][33];
  const int bx = blockIdx.x * 32, by = blockIdx.y * 32;
  const int tx = threadIdx.x & 31, ty = threadIdx.x >> 5;  // ty in 0..7
#pragma unroll
  for (int p = 0; p < 4; p++)
    tile[ty + p * 8][tx] = src[(size_t)(by + ty + p * 8) * NF + bx + tx];
  __syncthreads();
#pragma unroll
  for (int p = 0; p < 4; p++) {
    float x = tile[tx][ty + p * 8];
    __bf16 hi = (__bf16)x;
    size_t idx = (size_t)(bx + ty + p * 8) * NF + by + tx;
    dh[idx] = hi;
    dl[idx] = (__bf16)(x - (float)hi);
  }
}

// Row sums of (hi+lo)^2 : rg = sum_j D^2 (D symmetric => also column sums)
__global__ __launch_bounds__(256) void k_rowsum_bf(const __bf16* __restrict__ H,
                                                   const __bf16* __restrict__ L,
                                                   float* __restrict__ outv) {
  int row = blockIdx.x, t = threadIdx.x;
  const __bf16* ph = H + (size_t)row * NF;
  const __bf16* pl = L + (size_t)row * NF;
  float s = 0.f;
  for (int j = t; j < NF; j += 256) {
    float x = (float)ph[j] + (float)pl[j];
    s += x * x;
  }
  __shared__ float sh[256];
  sh[t] = s; __syncthreads();
  for (int o = 128; o > 0; o >>= 1) { if (t < o) sh[t] += sh[t + o]; __syncthreads(); }
  if (t == 0) outv[row] = sh[0];
}

// ---------------------------------------------------------------------------
// Deterministic two-stage reductions (no atomics -> bitwise replayable)
// ---------------------------------------------------------------------------
__global__ __launch_bounds__(256) void k_absmax_stage1(const float* __restrict__ src,
                                                       float* __restrict__ part) {
  float m = 0.f;
  for (size_t idx = (size_t)blockIdx.x * 256 + threadIdx.x; idx < N2;
       idx += (size_t)gridDim.x * 256)
    m = fmaxf(m, fabsf(src[idx]));
  __shared__ float sh[256];
  int t = threadIdx.x;
  sh[t] = m; __syncthreads();
  for (int o = 128; o > 0; o >>= 1) { if (t < o) sh[t] = fmaxf(sh[t], sh[t + o]); __syncthreads(); }
  if (t == 0) part[blockIdx.x] = sh[0];
}

__global__ __launch_bounds__(256) void k_max_stage2(const float* __restrict__ part,
                                                    int nparts, float* __restrict__ outsc) {
  float m = -INFINITY;
  for (int i = threadIdx.x; i < nparts; i += 256) m = fmaxf(m, part[i]);
  __shared__ float sh[256];
  int t = threadIdx.x;
  sh[t] = m; __syncthreads();
  for (int o = 128; o > 0; o >>= 1) { if (t < o) sh[t] = fmaxf(sh[t], sh[t + o]); __syncthreads(); }
  if (t == 0) *outsc = sh[0];
}

__global__ __launch_bounds__(256) void k_sum_stage2(const float* __restrict__ part,
                                                    int nparts, float* __restrict__ outsc) {
  float s = 0.f;
  for (int i = threadIdx.x; i < nparts; i += 256) s += part[i];
  __shared__ float sh[256];
  int t = threadIdx.x;
  sh[t] = s; __syncthreads();
  for (int o = 128; o > 0; o >>= 1) { if (t < o) sh[t] += sh[t + o]; __syncthreads(); }
  if (t == 0) *outsc = sh[0];
}

__global__ __launch_bounds__(256) void k_rowsum(const float* __restrict__ M,
                                                float* __restrict__ outv) {
  int row = blockIdx.x, t = threadIdx.x;
  const float* p = M + (size_t)row * NF;
  float s = 0.f;
  for (int j = t; j < NF; j += 256) s += p[j];
  __shared__ float sh[256];
  sh[t] = s; __syncthreads();
  for (int o = 128; o > 0; o >>= 1) { if (t < o) sh[t] += sh[t + o]; __syncthreads(); }
  if (t == 0) outv[row] = sh[0];
}

__global__ __launch_bounds__(256) void k_colsum(const float* __restrict__ M,
                                                float* __restrict__ outv) {
  int j = blockIdx.x * 256 + threadIdx.x;
  float s = 0.f;
  for (int r = 0; r < NF; ++r) s += M[(size_t)r * NF + j];
  outv[j] = s;
}

__global__ __launch_bounds__(256) void k_fill(float* __restrict__ p, float v, size_t n) {
  for (size_t idx = (size_t)blockIdx.x * 256 + threadIdx.x; idx < n;
       idx += (size_t)gridDim.x * 256)
    p[idx] = v;
}

// C_fused = 0.5*C_W/cwmax + 0.5*C_GW/gmax -> TL ; per-block max(C_fused)
__global__ __launch_bounds__(256) void k_fuse(const float* __restrict__ Cw,
                                              const float* __restrict__ Cgw,
                                              const float* __restrict__ sc,
                                              float* __restrict__ TL,
                                              float* __restrict__ part) {
  float cw = sc[0], g = sc[1];
  float icw = cw > 0.f ? 1.f / cw : 1.f;
  float ig = g > 0.f ? 1.f / g : 1.f;
  float m = -INFINITY;
  for (size_t idx = (size_t)blockIdx.x * 256 + threadIdx.x; idx < N2;
       idx += (size_t)gridDim.x * 256) {
    float cf = 0.5f * Cw[idx] * icw + 0.5f * Cgw[idx] * ig;
    TL[idx] = cf;
    m = fmaxf(m, cf);
  }
  __shared__ float sh[256];
  int t = threadIdx.x;
  sh[t] = m; __syncthreads();
  for (int o = 128; o > 0; o >>= 1) { if (t < o) sh[t] = fmaxf(sh[t], sh[t + o]); __syncthreads(); }
  if (t == 0) part[blockIdx.x] = sh[0];
}

// TL <- -(TL / fmax) / EPSILON
__global__ __launch_bounds__(256) void k_logk(float* __restrict__ TL,
                                              const float* __restrict__ sc) {
  float f = sc[2];
  float scale = (f > 0.f ? 1.f / f : 1.f) * (-1.f / EPSILON);
  for (size_t idx = (size_t)blockIdx.x * 256 + threadIdx.x; idx < N2;
       idx += (size_t)gridDim.x * 256)
    TL[idx] *= scale;
}

// ---------------------------------------------------------------------------
// Sinkhorn: single-pass online logsumexp (v_exp_f32 transcendental pipe)
// ---------------------------------------------------------------------------
__global__ __launch_bounds__(256) void k_uupd(const float* __restrict__ LK,
                                              const float* __restrict__ v,
                                              float* __restrict__ u) {
  int i = blockIdx.x, t = threadIdx.x;
  const float* row = LK + (size_t)i * NF;
  float m = -INFINITY, s = 0.f;
  for (int j = t; j < NF; j += 256) {
    float x = row[j] + v[j];
    if (x > m) { s = s * __expf(m - x) + 1.f; m = x; }
    else       { s += __expf(x - m); }
  }
  __shared__ float sm[256], ss[256];
  sm[t] = m; ss[t] = s; __syncthreads();
  for (int o = 128; o > 0; o >>= 1) {
    if (t < o) {
      float m1 = sm[t], s1 = ss[t], m2 = sm[t + o], s2 = ss[t + o];
      float M = fmaxf(m1, m2);
      ss[t] = s1 * __expf(m1 - M) + s2 * __expf(m2 - M);
      sm[t] = M;
    }
    __syncthreads();
  }
  if (t == 0) {
    float log_a = __logf(1.f / (float)NF + EPS_LOG);
    u[i] = log_a - (sm[0] + __logf(ss[0]));
  }
}

__global__ __launch_bounds__(256) void k_vupd(const float* __restrict__ LK,
                                              const float* __restrict__ u,
                                              float* __restrict__ v) {
  int j = blockIdx.x * 256 + threadIdx.x;  // one column/thread, coalesced rows
  float m = -INFINITY, s = 0.f;
  for (int r = 0; r < NF; ++r) {
    float x = LK[(size_t)r * NF + j] + u[r];
    if (x > m) { s = s * __expf(m - x) + 1.f; m = x; }
    else       { s += __expf(x - m); }
  }
  float log_b = __logf(1.f / (float)NF + EPS_LOG);
  v[j] = log_b - (m + __logf(s));
}

__global__ __launch_bounds__(256) void k_formP(const float* __restrict__ LK,
                                               const float* __restrict__ u,
                                               const float* __restrict__ v,
                                               float* __restrict__ P,
                                               float* __restrict__ part) {
  float s = 0.f;
  for (size_t idx = (size_t)blockIdx.x * 256 + threadIdx.x; idx < N2;
       idx += (size_t)gridDim.x * 256) {
    int i = (int)(idx >> 12);
    int j = (int)(idx & (NF - 1));
    float p = __expf(u[i] + LK[idx] + v[j]);
    P[idx] = p;
    s += p;
  }
  __shared__ float sh[256];
  int t = threadIdx.x;
  sh[t] = s; __syncthreads();
  for (int o = 128; o > 0; o >>= 1) { if (t < o) sh[t] += sh[t + o]; __syncthreads(); }
  if (t == 0) part[blockIdx.x] = sh[0];
}

__global__ __launch_bounds__(256) void k_scaleP(float* __restrict__ P,
                                                const float* __restrict__ sc) {
  float inv = 1.f / (sc[3] + EPS_LOG);
  for (size_t idx = (size_t)blockIdx.x * 256 + threadIdx.x; idx < N2;
       idx += (size_t)gridDim.x * 256)
    P[idx] *= inv;
}

__global__ __launch_bounds__(256) void k_final(const float* __restrict__ Cw,
                                               const float* __restrict__ Cgw,
                                               float* __restrict__ out2) {
  for (size_t idx = (size_t)blockIdx.x * 256 + threadIdx.x; idx < N2;
       idx += (size_t)gridDim.x * 256)
    out2[idx] = 0.5f * Cw[idx] + 0.5f * Cgw[idx];
}

// ---------------------------------------------------------------------------
extern "C" void kernel_launch(void* const* d_in, const int* in_sizes, int n_in,
                              void* d_out, int out_size, void* d_ws, size_t ws_size,
                              hipStream_t stream) {
  (void)in_sizes; (void)n_in; (void)out_size; (void)ws_size;
  const float* gen = (const float*)d_in[0];
  const float* rea = (const float*)d_in[1];
  float* out = (float*)d_out;
  float* P = out;               // coupling lives in first half of d_out
  float* Cfinal = out + N2;

  float* ws  = (float*)d_ws;
  float* Cw  = ws;              // f32 matrices
  float* Cgw = ws + N2;
  float* TLf = ws + 2 * N2;     // shared: T (= Dg@P), then C_fused, then log_K
  __bf16* DgH = (__bf16*)(ws + 3 * N2);  // split-bf16 matrices (half size)
  __bf16* DgL = DgH + N2;
  __bf16* DrH = DgL + N2;
  __bf16* DrL = DrH + N2;
  __bf16* XH  = DrL + N2;       // shared: P^T split, then T split
  __bf16* XL  = XH + N2;
  float* vec = (float*)(XL + N2);
  float *sqg = vec, *sqr = vec + NF, *rg = vec + 2 * NF, *rr = vec + 3 * NF,
        *mu = vec + 4 * NF, *nu = vec + 5 * NF, *uu = vec + 6 * NF,
        *vv = vec + 7 * NF;
  float* part = vec + 8 * NF;   // 1024 partials
  float* sc   = part + 1024;    // [0]=cwmax [1]=gmax [2]=fmax [3]=sumP

  dim3 distGrid(32, 16);        // (M/128) x (N/256), f32-WMMA NT GEMM
  dim3 bigGrid(32, 32);         // (M/128) x (N/128), split-bf16 GEMM
  dim3 trGrid(128, 128);        // 32x32 transpose tiles

  // Distance matrices (split-bf16 out) + Wasserstein cost (f32)
  k_rownorm<<<NF, 256, 0, stream>>>(gen, sqg);
  k_rownorm<<<NF, 256, 0, stream>>>(rea, sqr);
  k_gemm_nt_dist<<<distGrid, 256, 0, stream>>>(gen, gen, sqg, sqg, nullptr, DgH, DgL, 1);
  k_gemm_nt_dist<<<distGrid, 256, 0, stream>>>(rea, rea, sqr, sqr, nullptr, DrH, DrL, 1);
  k_gemm_nt_dist<<<distGrid, 256, 0, stream>>>(gen, rea, sqg, sqr, Cw, nullptr, nullptr, 0);
  k_absmax_stage1<<<1024, 256, 0, stream>>>(Cw, part);
  k_max_stage2<<<1, 256, 0, stream>>>(part, 1024, sc + 0);
  k_rowsum_bf<<<NF, 256, 0, stream>>>(DgH, DgL, rg);   // sum_j D_gen^2 rows
  k_rowsum_bf<<<NF, 256, 0, stream>>>(DrH, DrL, rr);   // symmetric => col sums
  k_fill<<<2048, 256, 0, stream>>>(P, 1.f / ((float)NF * (float)NF), N2);

  for (int it = 0; it < NUM_FGW; ++it) {
    k_rowsum<<<NF, 256, 0, stream>>>(P, mu);
    k_colsum<<<NF / 256, 256, 0, stream>>>(P, nu);
    // T = Dg @ P = Dg @ (P^T)^T  -> NT split-bf16 GEMM
    k_cvt_split_T<<<trGrid, 256, 0, stream>>>(P, XH, XL);
    k_gemm_bf16split<<<bigGrid, 256, 0, stream>>>(DgH, DgL, XH, XL, TLf,
                                                  nullptr, nullptr, nullptr,
                                                  nullptr, 0);
    // C_GW = rg*nu + mu*rr - 2 * T @ Dr   (Dr symmetric => NT form directly)
    k_cvt_split<<<2048, 256, 0, stream>>>(TLf, XH, XL);
    k_gemm_bf16split<<<bigGrid, 256, 0, stream>>>(XH, XL, DrH, DrL, Cgw,
                                                  rg, nu, mu, rr, 1);
    k_absmax_stage1<<<1024, 256, 0, stream>>>(Cgw, part);
    k_max_stage2<<<1, 256, 0, stream>>>(part, 1024, sc + 1);
    k_fuse<<<1024, 256, 0, stream>>>(Cw, Cgw, sc, TLf, part);
    k_max_stage2<<<1, 256, 0, stream>>>(part, 1024, sc + 2);
    k_logk<<<2048, 256, 0, stream>>>(TLf, sc);
    // Sinkhorn (log-stabilized), u0 = v0 = 0
    k_fill<<<16, 256, 0, stream>>>(uu, 0.f, (size_t)NF);
    k_fill<<<16, 256, 0, stream>>>(vv, 0.f, (size_t)NF);
    for (int s = 0; s < NUM_SINKHORN; ++s) {
      k_uupd<<<NF, 256, 0, stream>>>(TLf, vv, uu);
      k_vupd<<<NF / 256, 256, 0, stream>>>(TLf, uu, vv);
    }
    k_formP<<<1024, 256, 0, stream>>>(TLf, uu, vv, P, part);
    k_sum_stage2<<<1, 256, 0, stream>>>(part, 1024, sc + 3);
    k_scaleP<<<2048, 256, 0, stream>>>(P, sc);
  }
  k_final<<<2048, 256, 0, stream>>>(Cw, Cgw, Cfinal);
}
